// CausalSelfAttention_57415122813296
// MI455X (gfx1250) — matrix operand order repaired and checked by reading, not compiled, not run on hardware
//
#include <hip/hip_runtime.h>
#include <hip/hip_bf16.h>

// Problem constants (match reference)
#define B_   2
#define N_   2048
#define H_   16
#define D_   1024
#define HD_  64
#define F_   192   // 3*HD (split order: k, q, v)

typedef _Float16 f16;
typedef __attribute__((ext_vector_type(16))) _Float16 v16h;
typedef __attribute__((ext_vector_type(8)))  _Float16 v8h;
typedef __attribute__((ext_vector_type(8)))  float    v8f;

// ---- WMMA wrapper: D = A(16x32 f16) * B(32x16 f16) + C(16x16 f32) ----
__device__ __forceinline__ v8f wmma_f32(v16h a, v16h b, v8f c) {
  return __builtin_amdgcn_wmma_f32_16x16x32_f16(
      /*neg_a=*/false, a, /*neg_b=*/false, b,
      /*c_mod=*/(short)0, c, /*reuse_a=*/false, /*reuse_b=*/false);
}

// A-operand (16-bit 16x32 MxK): lane L<16 row=L holds K {k0..k0+7, k0+16..k0+23};
// lane L>=16 row=L-16 holds K {k0+8..k0+15, k0+24..k0+31}.
__device__ __forceinline__ v16h load_A16(const f16* __restrict__ base, int ld,
                                         int row0, int k0) {
  int lane = threadIdx.x & 31;
  const f16* p = base + (size_t)(row0 + (lane & 15)) * ld + k0 + ((lane >> 4) << 3);
  v8h lo = *(const v8h*)p;
  v8h hi = *(const v8h*)(p + 16);
  v16h r;
#pragma unroll
  for (int i = 0; i < 8; ++i) { r[i] = lo[i]; r[i + 8] = hi[i]; }
  return r;
}

// B-operand (16-bit 32x16 KxN), sourced from a row-major NxK (i.e. B^T) matrix:
// lane L<16 col N=L holds K {k0..k0+7, k0+16..k0+23}; lane L>=16 holds K {k0+8..k0+15, k0+24..k0+31}.
__device__ __forceinline__ v16h load_B16(const f16* __restrict__ base, int ld,
                                         int n0, int k0) {
  int lane = threadIdx.x & 31;
  const f16* p = base + (size_t)(n0 + (lane & 15)) * ld + k0 + ((lane >> 4) << 3);
  v8h lo = *(const v8h*)p;
  v8h hi = *(const v8h*)(p + 16);
  v16h r;
#pragma unroll
  for (int i = 0; i < 8; ++i) { r[i] = lo[i]; r[i + 8] = hi[i]; }
  return r;
}

// ---------------- Kernel 1: x f32 -> f16 ----------------
__global__ void cvt_x_kernel(const float* __restrict__ x, f16* __restrict__ xh, int n) {
  int i = blockIdx.x * blockDim.x + threadIdx.x;
  if (i < n) xh[i] = (f16)x[i];
}

// ---------------- Kernel 2: Wkqv[h][k][f] -> Wt[h][f][k] f16 ----------------
__global__ void pack_w_kernel(const float* __restrict__ W, f16* __restrict__ Wt) {
  int e = blockIdx.x * blockDim.x + threadIdx.x;
  if (e >= H_ * F_ * D_) return;
  int h = e / (F_ * D_);
  int r = e % (F_ * D_);
  int f = r / D_;
  int k = r % D_;
  Wt[((size_t)h * F_ + f) * D_ + k] = (f16)W[((size_t)h * D_ + k) * F_ + f];
}

// ---------------- Kernel 3: QKV projection GEMM (ping-pong pipelined) --------
// Each wave: 16 token rows x 64 proj cols; K-loop unrolled 2x (step 64) with
// two disjoint operand buffer sets so loads write directly into consuming
// registers (no rotation copies, no full loadcnt drain per step).
// The final-iteration over-prefetch reads into the following workspace region
// (memory-safe) and is never consumed.
// proj cols 0..63 -> K, 64..127 -> Q, 128..191 -> V (stored transposed).
__global__ __launch_bounds__(256) void qkv_gemm_kernel(
    const f16* __restrict__ xh, const f16* __restrict__ wt,
    const float* __restrict__ bias,
    f16* __restrict__ Qb, f16* __restrict__ Kb, f16* __restrict__ Vt) {
  int wave = (blockIdx.x << 3) + (threadIdx.x >> 5);   // 0..12287
  int rt = wave / 48;                                  // 0..255 (row tile of 16)
  int ct = wave % 48;                                  // head*3 + segment
  int b  = rt >> 7;
  int n0 = (rt & 127) << 4;
  int h   = ct / 3;
  int seg = ct % 3;          // 0=k 1=q 2=v
  int f0  = seg * 64;
  const f16* X = xh + (size_t)b * N_ * D_;
  const f16* W = wt + (size_t)h * F_ * D_;

  v8f acc[4] = {};

  // prologue: buffer set 0 <- k=0
  v16h a0 = load_A16(X, D_, n0, 0);
  v16h b0[4];
#pragma unroll
  for (int nt = 0; nt < 4; ++nt) b0[nt] = load_B16(W, D_, f0 + nt * 16, 0);

  for (int k = 0; k < D_; k += 64) {
    // buffer set 1 <- k+32 (10 b128 issued before any wait)
    v16h a1 = load_A16(X, D_, n0, k + 32);
    v16h b1[4];
#pragma unroll
    for (int nt = 0; nt < 4; ++nt) b1[nt] = load_B16(W, D_, f0 + nt * 16, k + 32);
    // compute k (waits only on buffer-0 loads issued one phase earlier)
#pragma unroll
    for (int nt = 0; nt < 4; ++nt) acc[nt] = wmma_f32(a0, b0[nt], acc[nt]);
    // buffer set 0 <- k+64 (last iteration: harmless over-read, never consumed)
    a0 = load_A16(X, D_, n0, k + 64);
#pragma unroll
    for (int nt = 0; nt < 4; ++nt) b0[nt] = load_B16(W, D_, f0 + nt * 16, k + 64);
    // compute k+32
#pragma unroll
    for (int nt = 0; nt < 4; ++nt) acc[nt] = wmma_f32(a1, b1[nt], acc[nt]);
  }

  int lane = threadIdx.x & 31;
  int rowAdd = (lane >> 4) << 3;     // D layout: rows i (lanes<16) / i+8 (lanes>=16)
  size_t bh = (size_t)(b * H_ + h);
#pragma unroll
  for (int nt = 0; nt < 4; ++nt) {
    int fc = nt * 16 + (lane & 15);  // 0..63 within segment
    float bv = bias[h * F_ + f0 + fc];
#pragma unroll
    for (int i = 0; i < 8; ++i) {
      int n = n0 + i + rowAdd;
      f16 v = (f16)(acc[nt][i] + bv);
      if (seg == 0)      Kb[(bh * N_ + n) * HD_ + fc] = v;
      else if (seg == 1) Qb[(bh * N_ + n) * HD_ + fc] = v;
      else               Vt[(bh * HD_ + fc) * N_ + n] = v;  // transposed
    }
  }
}

// ---------------- Kernel 4: causal flash attention ----------------
// Block = 4 waves (128 thr); wave w handles q rows [blockIdx.x*64 + w*16, +16).
// kv processed in blocks of 32 keys: 4 WMMAs for S, LDS relayout, 4 WMMAs for O.
// V-tile loads are hoisted before the softmax so their latency hides under the
// row-max/exp/reduction VALU work.
__global__ __launch_bounds__(128) void attn_kernel(
    const f16* __restrict__ Qg, const f16* __restrict__ Kg,
    const f16* __restrict__ Vtg, float* __restrict__ out) {
  __shared__ __align__(16) f16 plds[4][16 * 32];   // per-wave P staging tile
  int wave = threadIdx.x >> 5;
  int lane = threadIdx.x & 31;
  int qBase = blockIdx.x * 64 + wave * 16;
  int bh = blockIdx.y;
  int b = bh >> 4, h = bh & 15;
  const f16* Q  = Qg  + (size_t)bh * N_ * HD_;
  const f16* K  = Kg  + (size_t)bh * N_ * HD_;
  const f16* Vt = Vtg + (size_t)bh * HD_ * N_;

  v16h aq0 = load_A16(Q, HD_, qBase, 0);
  v16h aq1 = load_A16(Q, HD_, qBase, 32);

  float m[8], l[8];
#pragma unroll
  for (int i = 0; i < 8; ++i) { m[i] = -1e30f; l[i] = 0.f; }
  v8f acc[4] = {};
  int rowAdd = (lane >> 4) << 3;
  int col16  = lane & 15;
  f16* pw = &plds[wave][0];

  for (int j0 = 0; j0 <= qBase + 15; j0 += 32) {
    // issue V-tile loads early; consumed only after the softmax below
    v16h bv[4];
#pragma unroll
    for (int nt = 0; nt < 4; ++nt) bv[nt] = load_B16(Vt, N_, nt * 16, j0);

    // S = Q * K^T for 16x32 keys
    v8f s[2] = {};
#pragma unroll
    for (int t = 0; t < 2; ++t) {
      s[t] = wmma_f32(aq0, load_B16(K, HD_, j0 + t * 16, 0),  s[t]);
      s[t] = wmma_f32(aq1, load_B16(K, HD_, j0 + t * 16, 32), s[t]);
    }
    // scale + causal mask
#pragma unroll
    for (int i = 0; i < 8; ++i) {
      int row = qBase + i + rowAdd;
#pragma unroll
      for (int t = 0; t < 2; ++t) {
        int colk = j0 + t * 16 + col16;
        float v = s[t][i] * 0.125f;          // 1/sqrt(64)
        s[t][i] = (colk > row) ? -1e30f : v;
      }
    }
    // online softmax (row reductions across the 16-lane half-groups)
    float al[8];
#pragma unroll
    for (int i = 0; i < 8; ++i) {
      float rmax = fmaxf(s[0][i], s[1][i]);
#pragma unroll
      for (int off = 1; off < 16; off <<= 1)
        rmax = fmaxf(rmax, __shfl_xor(rmax, off, 32));
      float mn = fmaxf(m[i], rmax);
      float a  = __expf(m[i] - mn);
      float p0 = __expf(s[0][i] - mn);
      float p1 = __expf(s[1][i] - mn);
      float rs = p0 + p1;
#pragma unroll
      for (int off = 1; off < 16; off <<= 1)
        rs += __shfl_xor(rs, off, 32);
      l[i] = l[i] * a + rs;
      m[i] = mn;
      al[i] = a;
      // D-layout -> row-major LDS tile (16 rows x 32 kv)
      pw[(i + rowAdd) * 32 + col16]      = (f16)p0;
      pw[(i + rowAdd) * 32 + 16 + col16] = (f16)p1;
    }
#pragma unroll
    for (int nt = 0; nt < 4; ++nt)
#pragma unroll
      for (int i = 0; i < 8; ++i) acc[nt][i] *= al[i];

    // re-read P in A-operand layout from LDS (same-wave DS ops are in-order)
    const f16* pr = pw + (lane & 15) * 32 + ((lane >> 4) << 3);
    v8h lo = *(const v8h*)pr;
    v8h hi = *(const v8h*)(pr + 16);
    v16h ap;
#pragma unroll
    for (int i = 0; i < 8; ++i) { ap[i] = lo[i]; ap[i + 8] = hi[i]; }

    // O += P * V   (Vt rows = output cols, contiguous along kv)
#pragma unroll
    for (int nt = 0; nt < 4; ++nt)
      acc[nt] = wmma_f32(ap, bv[nt], acc[nt]);
  }

  // epilogue: O / l, write f32 out[b][n][h*64 + d]
#pragma unroll
  for (int nt = 0; nt < 4; ++nt) {
#pragma unroll
    for (int i = 0; i < 8; ++i) {
      int n = qBase + i + rowAdd;
      out[((size_t)b * N_ + n) * D_ + h * HD_ + nt * 16 + col16] = acc[nt][i] / l[i];
    }
  }
}

extern "C" void kernel_launch(void* const* d_in, const int* in_sizes, int n_in,
                              void* d_out, int out_size, void* d_ws, size_t ws_size,
                              hipStream_t stream) {
  (void)in_sizes; (void)n_in; (void)out_size; (void)ws_size;
  const float* x    = (const float*)d_in[0];   // [2,2048,1024]
  const float* W    = (const float*)d_in[1];   // [16,1024,192]
  const float* bias = (const float*)d_in[2];   // [16,192]
  float* out = (float*)d_out;                  // [2,2048,1024]

  // Workspace layout (f16): xh | Wt | Q | K | Vt  => ~38 MB total
  f16* xh = (f16*)d_ws;
  f16* Wt = xh + (size_t)B_ * N_ * D_;
  f16* Qb = Wt + (size_t)H_ * F_ * D_;
  f16* Kb = Qb + (size_t)B_ * H_ * N_ * HD_;
  f16* Vt = Kb + (size_t)B_ * H_ * N_ * HD_;

  int nx = B_ * N_ * D_;
  cvt_x_kernel<<<(nx + 255) / 256, 256, 0, stream>>>(x, xh, nx);
  int nw = H_ * F_ * D_;
  pack_w_kernel<<<(nw + 255) / 256, 256, 0, stream>>>(W, Wt);

  // 256 row-tiles * 48 (head,segment) tiles = 12288 waves, 8 waves/block
  qkv_gemm_kernel<<<1536, 256, 0, stream>>>(xh, Wt, bias, Qb, Kb, Vt);

  dim3 grid(N_ / 64, B_ * H_);
  attn_kernel<<<grid, 128, 0, stream>>>(Qb, Kb, Vt, out);
}
